// QKNormGatedCrossAttn_84473416778023
// MI455X (gfx1250) — matrix operand order repaired and checked by reading, not compiled, not run on hardware
//
#include <hip/hip_runtime.h>
#include <hip/hip_bf16.h>

typedef __attribute__((ext_vector_type(16))) _Float16 v16h;
typedef __attribute__((ext_vector_type(8)))  _Float16 v8h;
typedef __attribute__((ext_vector_type(8)))  float    v8f;
typedef __attribute__((ext_vector_type(4)))  float    v4f;

#define HEADS  16
#define DH     64
#define SQ     4096
#define SKV    512
#define DMODEL 1024
#define CENC   768

__device__ __forceinline__ v8f wmma_f16(v16h a, v16h b, v8f c) {
  // (neg_a, A, neg_b, B, c_mod, C, reuse_a, reuse_b)
  return __builtin_amdgcn_wmma_f32_16x16x32_f16(false, a, false, b, (short)0, c,
                                                false, false);
}

// Build a 16-half fragment from two contiguous 8-half runs: p[0..7], p[off2..off2+7].
// A-fragment (16x32 f16): lane m = L%16, k = (L/16)*16 + 0..15 contiguous (off2=8).
// B-fragment (32x16 f16): lane n = L%16, k = (L/16)*16 + 0..15 contiguous (off2=8).
__device__ __forceinline__ v16h load_frag(const _Float16* p, int off2) {
  v8h lo = *(const v8h*)p;
  v8h hi = *(const v8h*)(p + off2);
  v16h r;
#pragma unroll
  for (int i = 0; i < 8; ++i) { r[i] = lo[i]; r[8 + i] = hi[i]; }
  return r;
}

// gfx1250 async DMA: 16B global -> LDS, tracked by ASYNCcnt (no VGPR round-trip).
__device__ __forceinline__ void async_copy_b128(const _Float16* g, _Float16* l) {
  auto lp = (__attribute__((address_space(3))) _Float16*)l;  // 32-bit LDS offset
  asm volatile("global_load_async_to_lds_b128 %0, %1, off"
               :: "v"(lp), "v"(g)
               : "memory");
}
__device__ __forceinline__ void wait_async0() {
  asm volatile("s_wait_asynccnt 0x0" ::: "memory");
}

enum { OUT_QK = 0, OUT_VT = 1, OUT_F32 = 2 };

// C = A[M,K] @ W[K,1024] + bias. Tile 128x64, 8 waves; wave w owns rows w*16..+15 and
// ALL 64 columns (4 accumulators, 4 WMMA per 32-K step) -> wave-local L2-norm epilogue.
// OUT_QK : f16 -> [b, h, s, d]  (optionally L2-normalized rows per head)
// OUT_VT : f16 -> [b, h, d, kv] (transposed store for PV B-fragments)
// OUT_F32: f32 -> [m, 1024]
template <bool A_HALF, bool NORM, int OUTMODE>
__global__ __launch_bounds__(256) void gemm128(const void* __restrict__ Av,
                                               const float* __restrict__ W,
                                               const float* __restrict__ bias,
                                               void* __restrict__ outv,
                                               int M, int K, int Srows) {
  __shared__ __align__(16) _Float16 As[128][40];  // [m][k], padded stride
  __shared__ __align__(16) _Float16 Bs[64][40];   // [n][k] (transposed)

  const int t    = threadIdx.x;
  const int lane = t & 31, w = t >> 5;
  const int lm   = lane & 15, half = lane >> 4;
  const int m_base = blockIdx.x * 128;
  const int n_base = blockIdx.y * 64;

  v8f acc[4] = {};

  const int la_m = t >> 1, la_k = (t & 1) * 16;  // A stage: 128 rows x 32 k, 32B/thread
  const int lb_k = t >> 3, lb_n = (t & 7) * 8;   // B stage: 32 k x 64 n
  const int nk = K >> 5;

  for (int kb = 0; kb < nk; ++kb) {
    const int k0 = kb * 32;
    if (A_HALF) {
      // f16 A: DMA straight into LDS (ASYNCcnt), no conversion needed
      const _Float16* ga = (const _Float16*)Av + (size_t)(m_base + la_m) * K + k0 + la_k;
      async_copy_b128(ga,     &As[la_m][la_k]);
      async_copy_b128(ga + 8, &As[la_m][la_k + 8]);
      if (kb + 1 < nk) __builtin_prefetch(ga + 32, 0, 1);
    } else {
      // fp32 A: must pass through VALU for f32->f16 conversion
      const float* ga = (const float*)Av + (size_t)(m_base + la_m) * K + k0 + la_k;
      v4f a0 = *(const v4f*)ga;
      v4f a1 = *(const v4f*)(ga + 4);
      v4f a2 = *(const v4f*)(ga + 8);
      v4f a3 = *(const v4f*)(ga + 12);
      v8h h0, h1;
#pragma unroll
      for (int i = 0; i < 4; ++i) {
        h0[i] = (_Float16)a0[i]; h0[4 + i] = (_Float16)a1[i];
        h1[i] = (_Float16)a2[i]; h1[4 + i] = (_Float16)a3[i];
      }
      *(v8h*)&As[la_m][la_k]     = h0;
      *(v8h*)&As[la_m][la_k + 8] = h1;
      if (kb + 1 < nk) __builtin_prefetch(ga + 32, 0, 1);
    }
    {
      const float* gw = W + (size_t)(k0 + lb_k) * DMODEL + n_base + lb_n;
      v4f b0 = *(const v4f*)gw;
      v4f b1 = *(const v4f*)(gw + 4);
#pragma unroll
      for (int j = 0; j < 4; ++j) Bs[lb_n + j][lb_k] = (_Float16)b0[j];
#pragma unroll
      for (int j = 0; j < 4; ++j) Bs[lb_n + 4 + j][lb_k] = (_Float16)b1[j];
      if (kb + 1 < nk) __builtin_prefetch(gw + 32 * DMODEL, 0, 1);
    }
    if (A_HALF) wait_async0();  // async LDS writes must land before the barrier
    __syncthreads();
    v16h af = load_frag(&As[w * 16 + lm][half * 16], 8);
#pragma unroll
    for (int c = 0; c < 4; ++c) {
      v16h bf = load_frag(&Bs[c * 16 + lm][half * 16], 8);
      acc[c] = wmma_f16(af, bf, acc[c]);
    }
    __syncthreads();
  }

#pragma unroll
  for (int c = 0; c < 4; ++c) {
    const float bv = bias[n_base + c * 16 + lm];
#pragma unroll
    for (int r = 0; r < 8; ++r) acc[c][r] += bv;
  }

  float rn[8];
#pragma unroll
  for (int r = 0; r < 8; ++r) rn[r] = 1.0f;
  if (NORM) {
    // wave-local: wave holds all 64 head columns of its rows; butterfly over 16 lanes
    float ss[8];
#pragma unroll
    for (int r = 0; r < 8; ++r)
      ss[r] = acc[0][r] * acc[0][r] + acc[1][r] * acc[1][r] +
              acc[2][r] * acc[2][r] + acc[3][r] * acc[3][r];
#pragma unroll
    for (int mk = 1; mk < 16; mk <<= 1)
#pragma unroll
      for (int r = 0; r < 8; ++r) ss[r] += __shfl_xor(ss[r], mk, 32);
#pragma unroll
    for (int r = 0; r < 8; ++r)
      rn[r] = 1.0f / fmaxf(sqrtf(ss[r]), 1e-6f);
  }

  const int h = blockIdx.y;
#pragma unroll
  for (int r = 0; r < 8; ++r) {
    const int mg = m_base + w * 16 + half * 8 + r;  // C layout: row = r + 8*(lane/16)
    if (OUTMODE == OUT_QK) {
      _Float16* o = (_Float16*)outv;
      const int b = mg / Srows, s = mg % Srows;
      const size_t base = (((size_t)b * HEADS + h) * (size_t)Srows + s) * DH;
#pragma unroll
      for (int c = 0; c < 4; ++c)
        o[base + c * 16 + lm] = (_Float16)(acc[c][r] * rn[r]);
    } else if (OUTMODE == OUT_VT) {
      _Float16* o = (_Float16*)outv;
      const int b = mg / Srows, kv = mg % Srows;
      const size_t base = (((size_t)b * HEADS + h) * DH) * SKV + kv;
#pragma unroll
      for (int c = 0; c < 4; ++c)
        o[base + (size_t)(c * 16 + lm) * SKV] = (_Float16)acc[c][r];
    } else {
      float* o = (float*)outv;
#pragma unroll
      for (int c = 0; c < 4; ++c)
        o[(size_t)mg * DMODEL + n_base + c * 16 + lm] = acc[c][r];
    }
  }
}

__global__ void pool_kernel(const float* __restrict__ ehs, float* __restrict__ pooled) {
  const int i = blockIdx.x * blockDim.x + threadIdx.x;  // b*CENC + c
  if (i >= 4 * CENC) return;
  const int b = i / CENC, c = i % CENC;
  const float* p = ehs + (size_t)b * SKV * CENC + c;
  float s = 0.f;
  for (int j = 0; j < SKV; ++j) s += p[(size_t)j * CENC];
  pooled[i] = s * (1.0f / (float)SKV);
}

__global__ void gamma_kernel(const float* __restrict__ pooled,
                             const float* __restrict__ Wg,
                             const float* __restrict__ bg,
                             const float* __restrict__ head_gain,
                             float* __restrict__ gamma) {
  const int i = threadIdx.x;  // b*16 + h
  if (i >= 4 * HEADS) return;
  const int b = i / HEADS, h = i % HEADS;
  float s = bg[h];
  for (int c = 0; c < CENC; ++c) s += pooled[b * CENC + c] * Wg[c * HEADS + h];
  const float g1 = 1.0f / (1.0f + expf(-s));
  const float g2 = 1.0f / (1.0f + expf(-head_gain[h]));
  gamma[i] = g1 * g2;
}

// 4 waves, 64 queries/block, full Skv=512 in one shot. Scores f16 in 64KB dynamic LDS.
// Waves only touch their own 16 rows -> no barriers (same-wave LDS ordering via DScnt).
__global__ __launch_bounds__(128) void attn_kernel(const _Float16* __restrict__ qh,
                                                   const _Float16* __restrict__ kh,
                                                   const _Float16* __restrict__ vt,
                                                   const float* __restrict__ gamma,
                                                   _Float16* __restrict__ aout) {
  extern __shared__ _Float16 Sbuf[];  // [64][512] f16
  const int t    = threadIdx.x;
  const int lane = t & 31, w = t >> 5;
  const int lm   = lane & 15, half = lane >> 4;
  const int bq = blockIdx.x, h = blockIdx.y, b = blockIdx.z;
  const size_t bh = (size_t)b * HEADS + h;
  const _Float16* qp = qh + (bh * SQ + (size_t)bq * 64) * DH;
  const _Float16* kp = kh + bh * SKV * DH;
  const _Float16* vp = vt + bh * DH * SKV;

  // Q A-fragments (two d-chunks of 32) for this wave's 16 rows
  const int mrow = w * 16 + lm;
  v16h qa0 = load_frag(qp + (size_t)mrow * DH + half * 16, 8);
  v16h qa1 = load_frag(qp + (size_t)mrow * DH + 32 + half * 16, 8);

  // S = q @ k^T (raw, f16) -> LDS (K streams from L2: 64KB per (b,h))
  for (int tk = 0; tk < SKV / 16; ++tk) {
    const int kvb = tk * 16;
    const _Float16* kr = kp + (size_t)(kvb + lm) * DH;  // B lane: n = kv
    v16h kb0 = load_frag(kr + half * 16, 8);            // d 0..31
    v16h kb1 = load_frag(kr + 32 + half * 16, 8);       // d 32..63
    v8f c = {};
    c = wmma_f16(qa0, kb0, c);
    c = wmma_f16(qa1, kb1, c);
#pragma unroll
    for (int r = 0; r < 8; ++r)
      Sbuf[(size_t)(w * 16 + half * 8 + r) * SKV + kvb + lm] = (_Float16)c[r];
  }

  // softmax: lane handles row w*16+lm, columns half*256..+255; join halves via shfl_xor 16
  _Float16* rowp = Sbuf + (size_t)(w * 16 + lm) * SKV + half * 256;
  float mx = -3.0e38f;
  for (int j = 0; j < 256; j += 8) {
    v8h x = *(const v8h*)(rowp + j);
#pragma unroll
    for (int i = 0; i < 8; ++i) mx = fmaxf(mx, (float)x[i]);
  }
  mx = fmaxf(mx, __shfl_xor(mx, 16, 32));
  const float cexp = 0.125f * 1.44269504f;  // scale * log2(e)
  float sum = 0.f;
  for (int j = 0; j < 256; j += 8) {
    v8h x = *(const v8h*)(rowp + j);
    v8h y;
#pragma unroll
    for (int i = 0; i < 8; ++i) {
      const float p = exp2f(((float)x[i] - mx) * cexp);
      sum += p;
      y[i] = (_Float16)p;
    }
    *(v8h*)(rowp + j) = y;  // exp in place (unnormalized P)
  }
  sum += __shfl_xor(sum, 16, 32);
  const float g = (1.0f + gamma[bh]) / sum;  // fold 1/rowsum and head gate
  float rs[8];
#pragma unroll
  for (int r = 0; r < 8; ++r) rs[r] = __shfl(g, half * 8 + r, 32);

  // O = P @ V : A from LDS (own rows), B from V^T in global (L2-resident)
  v8f o0 = {}, o1 = {}, o2 = {}, o3 = {};
  const _Float16* prow = Sbuf + (size_t)(w * 16 + lm) * SKV;
  for (int cc = 0; cc < SKV / 32; ++cc) {
    v16h af  = load_frag(prow + cc * 32 + half * 16, 8);
    v16h bf0 = load_frag(vp + (size_t)(0 + lm) * SKV + cc * 32 + half * 16, 8);
    v16h bf1 = load_frag(vp + (size_t)(16 + lm) * SKV + cc * 32 + half * 16, 8);
    v16h bf2 = load_frag(vp + (size_t)(32 + lm) * SKV + cc * 32 + half * 16, 8);
    v16h bf3 = load_frag(vp + (size_t)(48 + lm) * SKV + cc * 32 + half * 16, 8);
    o0 = wmma_f16(af, bf0, o0);
    o1 = wmma_f16(af, bf1, o1);
    o2 = wmma_f16(af, bf2, o2);
    o3 = wmma_f16(af, bf3, o3);
  }

#pragma unroll
  for (int r = 0; r < 8; ++r) {
    const int s = bq * 64 + w * 16 + half * 8 + r;
    _Float16* orow = aout + ((size_t)b * SQ + s) * DMODEL + (size_t)h * DH;
    orow[0 + lm]  = (_Float16)(o0[r] * rs[r]);
    orow[16 + lm] = (_Float16)(o1[r] * rs[r]);
    orow[32 + lm] = (_Float16)(o2[r] * rs[r]);
    orow[48 + lm] = (_Float16)(o3[r] * rs[r]);
  }
}

extern "C" void kernel_launch(void* const* d_in, const int* in_sizes, int n_in,
                              void* d_out, int out_size, void* d_ws, size_t ws_size,
                              hipStream_t stream) {
  (void)in_sizes; (void)n_in; (void)out_size; (void)ws_size;
  const float* hs  = (const float*)d_in[0];
  const float* ehs = (const float*)d_in[1];
  const float* Wq  = (const float*)d_in[2];
  const float* bq  = (const float*)d_in[3];
  const float* Wk  = (const float*)d_in[4];
  const float* bk  = (const float*)d_in[5];
  const float* Wv  = (const float*)d_in[6];
  const float* bv  = (const float*)d_in[7];
  const float* Wo  = (const float*)d_in[8];
  const float* bo  = (const float*)d_in[9];
  const float* Wg  = (const float*)d_in[10];
  const float* bg  = (const float*)d_in[11];
  const float* hg  = (const float*)d_in[12];

  char* ws = (char*)d_ws;
  _Float16* q_h  = (_Float16*)(ws);                 // [4,16,4096,64] f16 : 32 MB
  _Float16* k_h  = (_Float16*)(ws + 33554432);      // [4,16,512,64]  f16 :  4 MB
  _Float16* v_t  = (_Float16*)(ws + 37748736);      // [4,16,64,512]  f16 :  4 MB
  _Float16* a_in = (_Float16*)(ws + 41943040);      // [4,4096,1024]  f16 : 32 MB
  float*    pooled = (float*)(ws + 75497472);       // [4,768] f32
  float*    gam    = (float*)(ws + 75509760);       // [4,16]  f32

  // QKV projections (f16 WMMA, f32 accum); Q/K fused bias + per-head L2 norm
  gemm128<false, true,  OUT_QK><<<dim3(4 * SQ / 128, HEADS), 256, 0, stream>>>(
      hs, Wq, bq, q_h, 4 * SQ, DMODEL, SQ);
  gemm128<false, true,  OUT_QK><<<dim3(4 * SKV / 128, HEADS), 256, 0, stream>>>(
      ehs, Wk, bk, k_h, 4 * SKV, CENC, SKV);
  gemm128<false, false, OUT_VT><<<dim3(4 * SKV / 128, HEADS), 256, 0, stream>>>(
      ehs, Wv, bv, v_t, 4 * SKV, CENC, SKV);

  // head gating
  pool_kernel<<<(4 * CENC + 255) / 256, 256, 0, stream>>>(ehs, pooled);
  gamma_kernel<<<1, 64, 0, stream>>>(pooled, Wg, bg, hg, gam);

  // attention (softmax + gate folded)
  attn_kernel<<<dim3(SQ / 64, HEADS, 4), 128, 64 * SKV * sizeof(_Float16), stream>>>(
      q_h, k_h, v_t, gam, a_in);

  // output projection -> fp32 (f16 A staged via async-to-LDS DMA)
  gemm128<true, false, OUT_F32><<<dim3(4 * SQ / 128, HEADS), 256, 0, stream>>>(
      a_in, Wo, bo, d_out, 4 * SQ, DMODEL, SQ);
}
